// CausalSelfAttention_42382737277037
// MI455X (gfx1250) — matrix operand order repaired and checked by reading, not compiled, not run on hardware
//
#include <hip/hip_runtime.h>
#include <hip/hip_bf16.h>

typedef _Float16 half_t;
typedef __attribute__((ext_vector_type(4)))  half_t v4h;
typedef __attribute__((ext_vector_type(8)))  half_t v8h;
typedef __attribute__((ext_vector_type(16))) half_t v16h;
typedef __attribute__((ext_vector_type(4)))  float  v4f;
typedef __attribute__((ext_vector_type(8)))  float  v8f;
typedef __attribute__((ext_vector_type(4)))  int    v4i;

#define DEVINL __device__ __forceinline__

#if __has_builtin(__builtin_amdgcn_global_load_async_to_lds_b128) && \
    __has_builtin(__builtin_amdgcn_s_wait_asynccnt)
#define USE_ASYNC_COPY 1
#else
#define USE_ASYNC_COPY 0
#endif

DEVINL v16h cat16(v8h a, v8h b) {
  return __builtin_shufflevector(a, b, 0,1,2,3,4,5,6,7,8,9,10,11,12,13,14,15);
}

// A-fragment (16x32, f16). src = element (0,0) of row-major region, ld = row
// stride (elems). Lane l: row = l&15; lanes 0-15 hold K={0..7,16..23},
// lanes 16-31 hold K={8..15,24..31}.
DEVINL v16h load_frag_a(const half_t* __restrict__ src, int ld, int lane) {
  int row   = lane & 15;
  int khalf = (lane >> 4) << 3;
  const half_t* p = src + (size_t)row * ld + khalf;
  v8h lo = *(const v8h*)(p);
  v8h hi = *(const v8h*)(p + 16);
  return cat16(lo, hi);
}

// B-fragment (32x16, f16) from K-contiguous storage: srcT row-major [N][K].
// Lane l: col = l&15; lanes 0-15 hold K={0..7,16..23}, lanes 16-31 hold K={8..15,24..31}.
DEVINL v16h load_frag_b(const half_t* __restrict__ srcT, int ld, int lane) {
  int col   = lane & 15;
  int kbase = (lane >> 4) << 3;
  const half_t* p = srcT + (size_t)col * ld + kbase;
  v8h lo = *(const v8h*)(p);
  v8h hi = *(const v8h*)(p + 16);
  return cat16(lo, hi);
}

DEVINL v8f wmma(v16h a, v16h b, v8f c) {
  return __builtin_amdgcn_wmma_f32_16x16x32_f16(false, a, false, b,
                                                (short)0, c, false, false);
}

// 16-byte global -> LDS copy. Async (ASYNCcnt) when the toolchain exposes the
// CDNA5 builtin, else plain VGPR staging. Builtin prototype (from hipcc
// diagnostics): (v4i AS1*, v4i AS3*, imm offset, imm cpol).
DEVINL void copy16_g2l(const half_t* g, half_t* l) {
#if USE_ASYNC_COPY
  __builtin_amdgcn_global_load_async_to_lds_b128(
      (__attribute__((address_space(1))) v4i*)const_cast<half_t*>(g),
      (__attribute__((address_space(3))) v4i*)l, 0, 0);
#else
  *(v8h*)l = *(const v8h*)g;
#endif
}

// ---------------------------------------------------------------------------
// f32 -> f16 convert (vectorized x4)
__global__ __launch_bounds__(256) void cvt_f16_kernel(
    const float* __restrict__ in, half_t* __restrict__ out, int n4) {
  int i = blockIdx.x * blockDim.x + threadIdx.x;
  int stride = gridDim.x * blockDim.x;
  const v4f* in4 = (const v4f*)in;
  v4h* out4 = (v4h*)out;
  for (; i < n4; i += stride)
    out4[i] = __builtin_convertvector(in4[i], v4h);
}

// in: f32 [K][N] -> out: f16 [N][K] (transpose + convert), K,N mult of 32
__global__ __launch_bounds__(256) void cvtT_kernel(
    const float* __restrict__ in, half_t* __restrict__ out, int K, int N) {
  __shared__ float tile[32][33];
  int n0 = blockIdx.x * 32;
  int k0 = blockIdx.y * 32;
  int tx = threadIdx.x & 31;
  int ty = threadIdx.x >> 5;  // 0..7
#pragma unroll
  for (int i = 0; i < 32; i += 8)
    tile[ty + i][tx] = in[(size_t)(k0 + ty + i) * N + n0 + tx];
  __syncthreads();
#pragma unroll
  for (int i = 0; i < 32; i += 8)
    out[(size_t)(n0 + ty + i) * K + k0 + tx] = (half_t)tile[tx][ty + i];
}

// ---------------------------------------------------------------------------
#define TM 128
#define TN 128
#define TK 32

#if USE_ASYNC_COPY
#define WAIT_ASYNC0() __builtin_amdgcn_s_wait_asynccnt(0)
#else
#define WAIT_ASYNC0()
#endif

// Shared GEMM mainloop: accumulates a 128x128 tile into acc[4][2] for this
// wave. Double-buffered async global->LDS pipeline (1 barrier / K-step) when
// available; classic 2-barrier staging otherwise.
#define GEMM_MAINLOOP(aG, bG, KDIM)                                           \
  {                                                                           \
    int lr = t >> 1;                                                          \
    int lc = (t & 1) << 4;                                                    \
    const half_t* aGp = (aG) + (size_t)lr * (KDIM) + lc;                      \
    const half_t* bGp = (bG) + (size_t)lr * (KDIM) + lc;                      \
    int soff = lr * TK + lc;                                                  \
    if (USE_ASYNC_COPY) {                                                     \
      copy16_g2l(aGp,     &As[0][soff]);                                      \
      copy16_g2l(aGp + 8, &As[0][soff + 8]);                                  \
      copy16_g2l(bGp,     &Bs[0][soff]);                                      \
      copy16_g2l(bGp + 8, &Bs[0][soff + 8]);                                  \
      for (int k0 = 0; k0 < (KDIM); k0 += TK) {                               \
        int buf = (k0 >> 5) & 1;                                              \
        WAIT_ASYNC0();                                                        \
        __syncthreads();                                                      \
        if (k0 + TK < (KDIM)) {                                               \
          int nb = buf ^ 1;                                                   \
          copy16_g2l(aGp + k0 + TK,     &As[nb][soff]);                       \
          copy16_g2l(aGp + k0 + TK + 8, &As[nb][soff + 8]);                   \
          copy16_g2l(bGp + k0 + TK,     &Bs[nb][soff]);                       \
          copy16_g2l(bGp + k0 + TK + 8, &Bs[nb][soff + 8]);                   \
        }                                                                     \
        v16h bf0 = load_frag_b(&Bs[buf][(wn * 32 + 0)  * TK], TK, lane);      \
        v16h bf1 = load_frag_b(&Bs[buf][(wn * 32 + 16) * TK], TK, lane);      \
        _Pragma("unroll")                                                     \
        for (int mi = 0; mi < 4; ++mi) {                                      \
          v16h af = load_frag_a(&As[buf][(wm * 64 + mi * 16) * TK], TK, lane);\
          acc[mi][0] = wmma(af, bf0, acc[mi][0]);                             \
          acc[mi][1] = wmma(af, bf1, acc[mi][1]);                             \
        }                                                                     \
      }                                                                       \
    } else {                                                                  \
      for (int k0 = 0; k0 < (KDIM); k0 += TK) {                               \
        v8h a0 = *(const v8h*)(aGp + k0);                                     \
        v8h a1 = *(const v8h*)(aGp + k0 + 8);                                 \
        v8h b0 = *(const v8h*)(bGp + k0);                                     \
        v8h b1 = *(const v8h*)(bGp + k0 + 8);                                 \
        __syncthreads();                                                      \
        *(v8h*)(&As[0][soff]) = a0; *(v8h*)(&As[0][soff + 8]) = a1;           \
        *(v8h*)(&Bs[0][soff]) = b0; *(v8h*)(&Bs[0][soff + 8]) = b1;           \
        __syncthreads();                                                      \
        v16h bf0 = load_frag_b(&Bs[0][(wn * 32 + 0)  * TK], TK, lane);        \
        v16h bf1 = load_frag_b(&Bs[0][(wn * 32 + 16) * TK], TK, lane);        \
        _Pragma("unroll")                                                     \
        for (int mi = 0; mi < 4; ++mi) {                                      \
          v16h af = load_frag_a(&As[0][(wm * 64 + mi * 16) * TK], TK, lane);  \
          acc[mi][0] = wmma(af, bf0, acc[mi][0]);                             \
          acc[mi][1] = wmma(af, bf1, acc[mi][1]);                             \
        }                                                                     \
      }                                                                       \
    }                                                                         \
  }

// QKV GEMM: xh[8192][1024] @ WaT[3072][1024] + bias -> scatter into Q, K, Vt.
__global__ __launch_bounds__(256) void qkv_gemm_kernel(
    const half_t* __restrict__ xh, const half_t* __restrict__ wT,
    const float* __restrict__ bias, half_t* __restrict__ Qh,
    half_t* __restrict__ Kh, half_t* __restrict__ Vt) {
  __shared__ alignas(16) half_t As[2][TM * TK];
  __shared__ alignas(16) half_t Bs[2][TM * TK];
  const int KDIM = 1024;
  int t = threadIdx.x, lane = t & 31, wave = t >> 5;
  int wm = wave & 1, wn = wave >> 1;
  int m0 = blockIdx.x * TM, n0 = blockIdx.y * TN;

  v8f acc[4][2] = {};
  GEMM_MAINLOOP(xh + (size_t)m0 * KDIM, wT + (size_t)n0 * KDIM, KDIM)

  int col16 = lane & 15;
  int rhalf = (lane >> 4) << 3;
#pragma unroll
  for (int ni = 0; ni < 2; ++ni) {
    int n = n0 + wn * 32 + ni * 16 + col16;
    float bv = bias[n];
    int part = n >> 10, rem = n & 1023;
    int h = rem >> 6, d = rem & 63;
#pragma unroll
    for (int mi = 0; mi < 4; ++mi) {
#pragma unroll
      for (int r = 0; r < 8; ++r) {
        int m = m0 + wm * 64 + mi * 16 + rhalf + r;
        int b = m >> 11, s = m & 2047;
        size_t bh = (size_t)(b * 16 + h);
        float v = acc[mi][ni][r] + bv;
        if (part == 0)       Qh[(bh * 2048 + s) * 64 + d] = (half_t)(v * 0.125f);
        else if (part == 1)  Kh[(bh * 2048 + s) * 64 + d] = (half_t)v;
        else                 Vt[(bh * 64 + d) * 2048 + s] = (half_t)v;
      }
    }
  }
}

// Output projection: Yh[8192][1024] @ WpT[1024][1024] + b_proj -> f32 out
__global__ __launch_bounds__(256) void proj_gemm_kernel(
    const half_t* __restrict__ yh, const half_t* __restrict__ wT,
    const float* __restrict__ bias, float* __restrict__ out) {
  __shared__ alignas(16) half_t As[2][TM * TK];
  __shared__ alignas(16) half_t Bs[2][TM * TK];
  const int KDIM = 1024;
  int t = threadIdx.x, lane = t & 31, wave = t >> 5;
  int wm = wave & 1, wn = wave >> 1;
  int m0 = blockIdx.x * TM, n0 = blockIdx.y * TN;

  v8f acc[4][2] = {};
  GEMM_MAINLOOP(yh + (size_t)m0 * KDIM, wT + (size_t)n0 * KDIM, KDIM)

  int col16 = lane & 15;
  int rhalf = (lane >> 4) << 3;
#pragma unroll
  for (int ni = 0; ni < 2; ++ni) {
    int n = n0 + wn * 32 + ni * 16 + col16;
    float bv = bias[n];
#pragma unroll
    for (int mi = 0; mi < 4; ++mi) {
#pragma unroll
      for (int r = 0; r < 8; ++r) {
        int m = m0 + wm * 64 + mi * 16 + rhalf + r;
        out[(size_t)m * 1024 + n] = acc[mi][ni][r] + bv;
      }
    }
  }
}

// ---------------------------------------------------------------------------
// Flash attention: grid (S/128, B*H), 8 waves/block, each wave owns 16 q rows.
// Scores computed transposed (S^T = K @ Q^T) so the probability fragment is
// directly the A operand of P@V. Online softmax; no LDS, no barriers.
__global__ __launch_bounds__(256) void attn_kernel(
    const half_t* __restrict__ Qh, const half_t* __restrict__ Kh,
    const half_t* __restrict__ Vt, half_t* __restrict__ Yh) {
  int lane = threadIdx.x & 31;
  int wave = threadIdx.x >> 5;
  int bh = blockIdx.y;
  int b = bh >> 4, h = bh & 15;
  int q0 = blockIdx.x * 128 + wave * 16;

  const half_t* Qp = Qh + (size_t)bh * 2048 * 64;
  const half_t* Kp = Kh + (size_t)bh * 2048 * 64;
  const half_t* Vp = Vt + (size_t)bh * 64 * 2048;

  // Q^T B-fragments (col = q, K-dim = d), reused all iterations
  v16h qb0 = load_frag_b(Qp + (size_t)q0 * 64, 64, lane);       // d 0..31
  v16h qb1 = load_frag_b(Qp + (size_t)q0 * 64 + 32, 64, lane);  // d 32..63

  v8f y[4] = {};
  const float NEG_INF = -__builtin_inff();
  float mrow = NEG_INF, lrow = 0.f;
  int qlane = q0 + (lane & 15);
  int khalf = (lane >> 4) << 3;
  int rbase = (lane >> 4) << 3;

  int kend = q0 + 16;
  for (int kt = 0; kt < kend; kt += 32) {
    if (kt + 32 < kend) {  // prefetch next K/V blocks into caches
      __builtin_prefetch(Kp + (size_t)(kt + 32) * 64, 0, 1);
      __builtin_prefetch(Vp + (size_t)(kt + 32), 0, 1);
    }
    v8f s0 = {}, s1 = {};
    {
      v16h ka = load_frag_a(Kp + (size_t)kt * 64, 64, lane);
      s0 = wmma(ka, qb0, s0);
      ka = load_frag_a(Kp + (size_t)kt * 64 + 32, 64, lane);
      s0 = wmma(ka, qb1, s0);
      ka = load_frag_a(Kp + (size_t)(kt + 16) * 64, 64, lane);
      s1 = wmma(ka, qb0, s1);
      ka = load_frag_a(Kp + (size_t)(kt + 16) * 64 + 32, 64, lane);
      s1 = wmma(ka, qb1, s1);
    }
    // causal mask + per-q block max (q = lane&15; keys striped over VGPRs)
    float bm = NEG_INF;
#pragma unroll
    for (int r = 0; r < 8; ++r) {
      int key0 = kt + khalf + r;
      float v0 = (key0 <= qlane) ? s0[r] : NEG_INF;
      float v1 = (key0 + 16 <= qlane) ? s1[r] : NEG_INF;
      s0[r] = v0; s1[r] = v1;
      bm = fmaxf(bm, fmaxf(v0, v1));
    }
    bm = fmaxf(bm, __shfl_xor(bm, 16, 32));
    float mnew = fmaxf(mrow, bm);
    float alpha = __expf(mrow - mnew);
    float ls = 0.f;
    v16h p;
#pragma unroll
    for (int r = 0; r < 8; ++r) {
      float p0 = __expf(s0[r] - mnew);
      float p1 = __expf(s1[r] - mnew);
      ls += p0 + p1;
      p[r] = (half_t)p0;       // A-frag elems 0..7  : keys khalf+0..7
      p[r + 8] = (half_t)p1;   // A-frag elems 8..15 : keys 16+khalf+0..7
    }
    lrow = lrow * alpha + ls + __shfl_xor(ls, 16, 32);
    mrow = mnew;
#pragma unroll
    for (int r = 0; r < 8; ++r) {
      float ar = __shfl(alpha, rbase + r, 32);
      y[0][r] *= ar; y[1][r] *= ar; y[2][r] *= ar; y[3][r] *= ar;
    }
#pragma unroll
    for (int f = 0; f < 4; ++f) {
      v16h vb = load_frag_b(Vp + (size_t)(f * 16) * 2048 + kt, 2048, lane);
      y[f] = wmma(p, vb, y[f]);
    }
  }

  float inv = 1.0f / lrow;  // valid for q = lane&15
  int dcol = lane & 15;
#pragma unroll
  for (int r = 0; r < 8; ++r) {
    float ir = __shfl(inv, rbase + r, 32);
    int q = q0 + rbase + r;
    size_t base = ((size_t)b * 2048 + q) * 1024 + h * 64;
#pragma unroll
    for (int f = 0; f < 4; ++f)
      Yh[base + f * 16 + dcol] = (half_t)(y[f][r] * ir);
  }
}

// ---------------------------------------------------------------------------
extern "C" void kernel_launch(void* const* d_in, const int* in_sizes, int n_in,
                              void* d_out, int out_size, void* d_ws,
                              size_t ws_size, hipStream_t stream) {
  (void)in_sizes; (void)n_in; (void)out_size; (void)ws_size;
  const float* x      = (const float*)d_in[0];
  const float* W_attn = (const float*)d_in[1];
  const float* b_attn = (const float*)d_in[2];
  const float* W_proj = (const float*)d_in[3];
  const float* b_proj = (const float*)d_in[4];
  float* out = (float*)d_out;

  char* ws = (char*)d_ws;
  const size_t XH_B  = (size_t)8192 * 1024 * 2;    // 16 MB
  const size_t WAT_B = (size_t)3072 * 1024 * 2;    // 6 MB
  const size_t WPT_B = (size_t)1024 * 1024 * 2;    // 2 MB
  const size_t QKV_B = (size_t)64 * 2048 * 64 * 2; // 16 MB each

  half_t* xh  = (half_t*)(ws);
  half_t* WaT = (half_t*)(ws + XH_B);
  half_t* WpT = (half_t*)(ws + XH_B + WAT_B);
  half_t* Qh  = (half_t*)(ws + XH_B + WAT_B + WPT_B);
  half_t* Kh  = (half_t*)(ws + XH_B + WAT_B + WPT_B + QKV_B);
  half_t* Vt  = (half_t*)(ws + XH_B + WAT_B + WPT_B + 2 * QKV_B);
  half_t* Yh  = (half_t*)(ws + XH_B + WAT_B + WPT_B + 3 * QKV_B);

  cvt_f16_kernel<<<2048, 256, 0, stream>>>(x, xh, 8192 * 1024 / 4);
  cvtT_kernel<<<dim3(3072 / 32, 1024 / 32), 256, 0, stream>>>(W_attn, WaT, 1024, 3072);
  cvtT_kernel<<<dim3(1024 / 32, 1024 / 32), 256, 0, stream>>>(W_proj, WpT, 1024, 1024);
  qkv_gemm_kernel<<<dim3(8192 / TM, 3072 / TN), 256, 0, stream>>>(
      xh, WaT, b_attn, Qh, Kh, Vt);
  attn_kernel<<<dim3(2048 / 128, 64), 256, 0, stream>>>(Qh, Kh, Vt, Yh);
  proj_gemm_kernel<<<dim3(8192 / TM, 1024 / TN), 256, 0, stream>>>(
      Yh, WpT, b_proj, out);
}
